// AdvancedHomeostaticCell_18769007084302
// MI455X (gfx1250) — matrix operand order repaired and checked by reading, not compiled, not run on hardware
//
#include <hip/hip_runtime.h>

// ---------- types ----------
typedef __attribute__((ext_vector_type(16))) _Float16 v16h;
typedef __attribute__((ext_vector_type(8)))  _Float16 v8h;
typedef __attribute__((ext_vector_type(8)))  float    v8f;
typedef __attribute__((ext_vector_type(4)))  unsigned int v4u;
typedef __attribute__((ext_vector_type(8)))  int      v8i;
typedef __attribute__((ext_vector_type(4)))  int      v4i;

#define D 128

// ---------- fast transcendentals (hw exp2 / rcp / rsq) ----------
__device__ __forceinline__ float fast_sigmoid(float x) {
    float e = __builtin_amdgcn_exp2f(-1.44269504f * x);
    return __builtin_amdgcn_rcpf(1.0f + e);
}
__device__ __forceinline__ float fast_tanh(float x) {
    // tanh(x) = 2*sigmoid(2x) - 1
    float e = __builtin_amdgcn_exp2f(-2.88539008f * x);
    return 2.0f * __builtin_amdgcn_rcpf(1.0f + e) - 1.0f;
}

// Build a 16-bit A-fragment (16x32) chunk from 16 f32 values:
// elems [0..7]  -> VGPRs 0..3  (K = 32c + 16h .. +7)
// elems [8..15] -> VGPRs 4..7  (K = 32c + 16h + 8 .. +15)
__device__ __forceinline__ v16h make_afrag(float4 p0, float4 p1, float4 q0, float4 q1) {
    v16h a;
    a[0]=(_Float16)p0.x;  a[1]=(_Float16)p0.y;  a[2]=(_Float16)p0.z;  a[3]=(_Float16)p0.w;
    a[4]=(_Float16)p1.x;  a[5]=(_Float16)p1.y;  a[6]=(_Float16)p1.z;  a[7]=(_Float16)p1.w;
    a[8]=(_Float16)q0.x;  a[9]=(_Float16)q0.y;  a[10]=(_Float16)q0.z; a[11]=(_Float16)q0.w;
    a[12]=(_Float16)q1.x; a[13]=(_Float16)q1.y; a[14]=(_Float16)q1.z; a[15]=(_Float16)q1.w;
    return a;
}

// ---------- prep: fuse + cast weights to f16 blob [Wo|Wi|Wa=Wslow+Wfast|Wf_x], fbias ----------
__global__ void ahc_prep(const float* __restrict__ Wslow, const float* __restrict__ Wfast,
                         const float* __restrict__ Wi,    const float* __restrict__ Wf,
                         const float* __restrict__ Wfb,   const float* __restrict__ hprev,
                         const float* __restrict__ Wo,
                         _Float16* __restrict__ wblob, float* __restrict__ fbias) {
    const int n = threadIdx.x;          // 0..127, one output neuron per thread
    #pragma unroll 4
    for (int k = 0; k < D; ++k) {
        wblob[0*16384 + n*D + k] = (_Float16)Wo[n*D + k];
        wblob[1*16384 + n*D + k] = (_Float16)Wi[n*D + k];
        wblob[2*16384 + n*D + k] = (_Float16)(Wslow[n*D + k] + Wfast[n*D + k]);
        wblob[3*16384 + n*D + k] = (_Float16)Wf[n*256 + k];            // x-part of Wf
    }
    float fb = Wfb[n];                                                  // hp-part folded into bias
    for (int k = 0; k < D; ++k) fb += Wf[n*256 + 128 + k] * hprev[k];
    fbias[n] = fb;
}

// ---------- main fused cell kernel: 256 threads = 8 waves, 16 rows/wave ----------
__global__ __launch_bounds__(256) void ahc_cell(
    const float* __restrict__ x,      const float* __restrict__ hprev,
    const float* __restrict__ Wslow_b,const float* __restrict__ Wi_b,
    const float* __restrict__ Wo_b,   const float* __restrict__ ln_g,
    const float* __restrict__ ln_b,   const _Float16* __restrict__ wblob,
    const float* __restrict__ fbias,  float* __restrict__ out)
{
    // 128 KB LDS: [Wo 32KB][Wi 32KB][Wa 32KB][Wf 32KB]; the Wi/Wa/Wf region is
    // reused after stage 1 as per-wave scratch (8 x 12KB) for h_raw f16 / out f32.
    __shared__ __attribute__((aligned(128))) _Float16 smem[65536];

    const int lane = threadIdx.x & 31;
    const int wave = threadIdx.x >> 5;
    const int h    = lane >> 4;          // lane half (A: K-sub-block, B: K-half, C: row-half)
    const int n0   = lane & 15;          // A: row, B/C: column within 16-tile
    const int row0 = blockIdx.x * 128 + wave * 16;

    // ---- TDM: DMA the 128KB weight blob into LDS (wave 0 only; EXEC ignored by TDM) ----
    if (threadIdx.x < 32) {
        unsigned int lds_off = (unsigned int)(unsigned long long)(uintptr_t)(&smem[0]);
        unsigned long long ga = (unsigned long long)(uintptr_t)wblob;
        v4u g0;
        g0[0] = 1u;                                             // count=1, user descriptor
        g0[1] = lds_off;                                        // lds_addr
        g0[2] = (unsigned int)ga;                               // global_addr[31:0]
        g0[3] = (unsigned int)((ga >> 32) & 0x1FFFFFFu) | 0x80000000u; // addr[56:32] | type=2
        v8i g1;
        g1[0] = (int)(3u << 16);         // workgroup_mask=0, data_size=3 (8B units)
        g1[1] = (int)(16384u << 16);     // tensor_dim0[15:0]=16384 (128KB / 8B)
        g1[2] = (int)(1u << 16);         // tensor_dim0 hi=0, tensor_dim1=1
        g1[3] = (int)(16384u << 16);     // tile_dim0=16384
        g1[4] = 1;                       // tile_dim1=1, tile_dim2=0
        g1[5] = 16384;                   // tensor_dim0_stride lo32
        g1[6] = 0;                       // stride hi / dim1 stride lo
        g1[7] = 0;
        v4i gz4 = {0, 0, 0, 0};          // groups 2/3 unused (<=2D tensor)
        v8i gz8 = {0, 0, 0, 0, 0, 0, 0, 0};
        __builtin_amdgcn_tensor_load_to_lds(g0, g1, gz4, gz4, gz8, 0);
        __builtin_amdgcn_s_wait_tensorcnt(0);
    }
    __syncthreads();

    // ---- build 4 A-fragments of this wave's 16 x-rows (f32 -> f16 in regs) ----
    v16h xa[4];
    {
        const float* xr = x + (size_t)(row0 + n0) * D;
        #pragma unroll
        for (int c = 0; c < 4; ++c) {
            const int klo = 32*c + 16*h;
            float4 p0 = *(const float4*)(xr + klo);
            float4 p1 = *(const float4*)(xr + klo + 4);
            float4 q0 = *(const float4*)(xr + klo + 8);
            float4 q1 = *(const float4*)(xr + klo + 12);
            xa[c] = make_afrag(p0, p1, q0, q1);
        }
    }

    const _Float16* Wo16 = smem;             // persists whole kernel
    const _Float16* Wi16 = smem + 16384;
    const _Float16* Wa16 = smem + 32768;
    const _Float16* Wf16 = smem + 49152;

    // ---- stage 1: three parallel GEMMs + gates; h_raw kept in 64 f32 regs ----
    v8f hr[8];
    #pragma unroll
    for (int j = 0; j < 8; ++j) {
        const int n = 16*j + n0;
        const int bidx = n * D + 16*h;       // lane's contiguous 16-f16 B chunk base
        v8f ai = {}, af = {}, as = {};
        #pragma unroll
        for (int c = 0; c < 4; ++c) {
            v16h bi = *(const v16h*)(Wi16 + bidx + 32*c);
            ai = __builtin_amdgcn_wmma_f32_16x16x32_f16(false, xa[c], false, bi, (short)0, ai, false, false);
            v16h ba = *(const v16h*)(Wa16 + bidx + 32*c);
            as = __builtin_amdgcn_wmma_f32_16x16x32_f16(false, xa[c], false, ba, (short)0, as, false, false);
            v16h bf = *(const v16h*)(Wf16 + bidx + 32*c);
            af = __builtin_amdgcn_wmma_f32_16x16x32_f16(false, xa[c], false, bf, (short)0, af, false, false);
        }
        const float wib = Wi_b[n], wsb = Wslow_b[n], fb = fbias[n], hp = hprev[n];
        #pragma unroll
        for (int r = 0; r < 8; ++r) {
            float it = fast_sigmoid(ai[r] + wib);
            float ft = fast_sigmoid(af[r] + fb);
            hr[j][r] = it * (as[r] + wsb) + ft * hp;
        }
    }

    // ---- all waves done reading Wi/Wa/Wf -> reuse that LDS as per-wave scratch ----
    __syncthreads();
    _Float16* sc16 = smem + 16384 + wave * 6144;   // 12KB/wave, f16 view, row stride 136
    #pragma unroll
    for (int j = 0; j < 8; ++j)
        #pragma unroll
        for (int r = 0; r < 8; ++r)
            sc16[(r + 8*h) * 136 + 16*j + n0] = (_Float16)hr[j][r];

    // ---- rebuild A-fragments from h_raw f16 (same-wave LDS is in-order) ----
    v16h ha[4];
    #pragma unroll
    for (int c = 0; c < 4; ++c) {
        const _Float16* pr = sc16 + n0 * 136 + 32*c + 16*h;
        v8h lo = *(const v8h*)pr;
        v8h hi = *(const v8h*)(pr + 8);
        v16h a;
        #pragma unroll
        for (int t = 0; t < 8; ++t) { a[t] = lo[t]; a[8+t] = hi[t]; }
        ha[c] = a;
    }

    // ---- stage 2: o-gate GEMM + h_out = o * tanh(h_raw) (overwrite hr) ----
    #pragma unroll
    for (int j = 0; j < 8; ++j) {
        const int n = 16*j + n0;
        const int bidx = n * D + 16*h;
        v8f ao = {};
        #pragma unroll
        for (int c = 0; c < 4; ++c) {
            v16h bo = *(const v16h*)(Wo16 + bidx + 32*c);
            ao = __builtin_amdgcn_wmma_f32_16x16x32_f16(false, ha[c], false, bo, (short)0, ao, false, false);
        }
        const float wob = Wo_b[n];
        #pragma unroll
        for (int r = 0; r < 8; ++r) {
            float ot = fast_sigmoid(ao[r] + wob);
            hr[j][r] = ot * fast_tanh(hr[j][r]);
        }
    }

    // ---- LayerNorm: per-register row sums + xor-butterfly within 16-lane halves ----
    float s[8], ss[8];
    #pragma unroll
    for (int r = 0; r < 8; ++r) { s[r] = 0.0f; ss[r] = 0.0f; }
    #pragma unroll
    for (int j = 0; j < 8; ++j)
        #pragma unroll
        for (int r = 0; r < 8; ++r) { float v = hr[j][r]; s[r] += v; ss[r] += v * v; }
    #pragma unroll
    for (int m = 1; m <= 8; m <<= 1) {
        #pragma unroll
        for (int r = 0; r < 8; ++r) {
            s[r]  += __shfl_xor(s[r],  m, 32);
            ss[r] += __shfl_xor(ss[r], m, 32);
        }
    }
    float mu[8], rstd[8];
    #pragma unroll
    for (int r = 0; r < 8; ++r) {
        mu[r] = s[r] * (1.0f / 128.0f);
        float var = ss[r] * (1.0f / 128.0f) - mu[r] * mu[r];
        rstd[r] = __builtin_amdgcn_rsqf(var + 1e-5f);
    }

    // ---- normalize, transpose through LDS, coalesced f32 stores ----
    float* scf = (float*)smem + 8192 + wave * 3072;   // f32 view of this wave's scratch
    #pragma unroll
    for (int j = 0; j < 8; ++j) {
        const int n = 16*j + n0;
        const float g = ln_g[n], bb = ln_b[n];
        #pragma unroll
        for (int r = 0; r < 8; ++r)
            scf[(r + 8*h) * 136 + n] = (hr[j][r] - mu[r]) * rstd[r] * g + bb;
    }
    __builtin_amdgcn_wave_barrier();   // keep store->load order across scheduler
    float* op = out + (size_t)row0 * D + lane * 4;
    #pragma unroll
    for (int t = 0; t < 16; ++t) {
        float4 v = *(const float4*)(scf + t * 136 + lane * 4);
        *(float4*)(op + (size_t)t * D) = v;
    }
}

// ---------- launch ----------
extern "C" void kernel_launch(void* const* d_in, const int* in_sizes, int n_in,
                              void* d_out, int out_size, void* d_ws, size_t ws_size,
                              hipStream_t stream) {
    const float* x       = (const float*)d_in[0];
    const float* hprev   = (const float*)d_in[1];
    const float* Wslow_w = (const float*)d_in[2];
    const float* Wslow_b = (const float*)d_in[3];
    const float* Wfast_w = (const float*)d_in[4];
    const float* Wi_w    = (const float*)d_in[5];
    const float* Wi_b    = (const float*)d_in[6];
    const float* Wf_w    = (const float*)d_in[7];
    const float* Wf_b    = (const float*)d_in[8];
    const float* Wo_w    = (const float*)d_in[9];
    const float* Wo_b    = (const float*)d_in[10];
    const float* ln_g    = (const float*)d_in[11];
    const float* ln_b    = (const float*)d_in[12];
    float* out = (float*)d_out;

    _Float16* wblob = (_Float16*)d_ws;                    // 128KB f16 blob
    float*    fbias = (float*)((char*)d_ws + 131072);     // 512B

    const int B = in_sizes[0] / D;                        // 262144 -> 2048 blocks of 128 rows
    ahc_prep<<<1, 128, 0, stream>>>(Wslow_w, Wfast_w, Wi_w, Wf_w, Wf_b, hprev, Wo_w, wblob, fbias);
    ahc_cell<<<B / 128, 256, 0, stream>>>(x, hprev, Wslow_b, Wi_b, Wo_b, ln_g, ln_b,
                                          wblob, fbias, out);
}
